// OptionsMamba2_13804024889694
// MI455X (gfx1250) — compile-verified, matrix operand order checked by reading, not run on hardware
//
#include <hip/hip_runtime.h>
#include <utility>

// ---------------- problem dims ----------------
constexpr int kB   = 16;
constexpr int kT   = 512;
constexpr int kDIN = 256;
constexpr int kDM  = 768;
constexpr int kL   = 4;
constexpr int kS   = 64;
constexpr int kDC  = 4;
constexpr int kDI  = 1536;          // DM * EXP
constexpr int kNT  = kB * kT;       // 8192 rows

// ---------------- types ----------------
typedef __bf16    bf16x16 __attribute__((ext_vector_type(16)));
typedef float     f32x8   __attribute__((ext_vector_type(8)));
typedef unsigned  uint32x4 __attribute__((ext_vector_type(4)));
typedef int       int32x4  __attribute__((ext_vector_type(4)));
typedef int       int32x8  __attribute__((ext_vector_type(8)));

// TDM availability (device pass only; host pass takes fallback)
#if defined(__gfx1250__) && __has_builtin(__builtin_amdgcn_tensor_load_to_lds)
#define HAVE_TDM 1
#else
#define HAVE_TDM 0
#endif

__device__ __forceinline__ unsigned short f2bf1(float a) {
    unsigned u = __float_as_uint(a);
    return (unsigned short)((u + (0x7FFFu + ((u >> 16) & 1u))) >> 16);
}
__device__ __forceinline__ unsigned pack_bf2(float a, float b) {
    return (unsigned)f2bf1(a) | ((unsigned)f2bf1(b) << 16);
}

__device__ __forceinline__ void wait_tensorcnt0() {
#if __has_builtin(__builtin_amdgcn_s_wait_tensorcnt)
    __builtin_amdgcn_s_wait_tensorcnt(0);
#else
    asm volatile("s_wait_tensorcnt 0x0" ::: "memory");
#endif
}

#if HAVE_TDM
// Issue one TDM 2-D tile load: 128 rows x 32 bf16 elems, row stride ld elems,
// into LDS at `sbuf` with 4-DWORD padding after every 16 DWORDs (80B rows).
__device__ __forceinline__ void tdm_load_tile(const unsigned short* g, int ld,
                                              unsigned row0, int k0, void* sbuf) {
    unsigned long long ga = (unsigned long long)g + 2ull * ((unsigned long long)row0 * (unsigned)ld + (unsigned)k0);
    unsigned ldsOff = (unsigned)(unsigned long long)sbuf;  // low 32 bits = LDS byte offset

    uint32x4 g0;
    g0.x = 1u;                                             // count=1, user descriptor
    g0.y = ldsOff;                                         // lds_addr
    g0.z = (unsigned)ga;                                   // global_addr[31:0]
    g0.w = (unsigned)((ga >> 32) & 0x1FFFFFFu) | 0x80000000u;  // global_addr[56:32] | type=2

    int32x8 g1;
    g1[0] = (int)((1u << 16)            // data_size = 2 bytes
                | (1u << 20)            // pad_enable
                | (3u << 22)            // pad_interval: 16 DWORDs
                | (3u << 25));          // pad_amount: 4 DWORDs
    g1[1] = (int)(((unsigned)ld & 0xFFFFu) << 16);         // tensor_dim0[15:0]
    g1[2] = (int)(((unsigned)ld >> 16) & 0xFFFFu);         // tensor_dim0[31:16], tensor_dim1[15:0]=0
    g1[3] = (int)(16u | (32u << 16));                      // tensor_dim1[31:16]=16 (=1M rows), tile_dim0=32
    g1[4] = 128;                                           // tile_dim1=128, tile_dim2=0
    g1[5] = ld;                                            // tensor_dim0_stride[31:0]
    g1[6] = 0;                                             // stride0[47:32]=0, stride1[15:0]=0
    g1[7] = 0;

    int32x4 z4 = {0, 0, 0, 0};
#if __clang_major__ >= 23
    int32x8 z8 = {0, 0, 0, 0, 0, 0, 0, 0};
    __builtin_amdgcn_tensor_load_to_lds(g0, g1, z4, z4, z8, 0);
#else
    __builtin_amdgcn_tensor_load_to_lds(g0, g1, z4, z4, 0);
#endif
}
#endif  // HAVE_TDM

// Load one 16x32 bf16 fragment from LDS (row-major, padded stride 40 bf16 = 20 dwords).
__device__ __forceinline__ bf16x16 load_frag(const unsigned* smem, int rowBase20, int lh) {
    union { bf16x16 v; unsigned u[8]; } f;
#pragma unroll
    for (int j = 0; j < 8; ++j)
        f.u[j] = smem[rowBase20 + ((j >> 2) << 3) + (lh << 2) + (j & 3)];
    return f.v;
}

// ---------------- GEMM: C[M,N] = A[M,K] @ W[N,K]^T (+bias) (+res) ----------
// bf16 operands pre-packed in memory. Tile 128x128x32; 8 waves, wave tile
// 64x32 = 4x2 wmma. Requires M%128==0, N%128==0, K%32==0 (true everywhere).
template <bool HAS_BIAS, bool ADD_RES>
__global__ __launch_bounds__(256) void gemm_bf16(
    const unsigned short* __restrict__ A, const unsigned short* __restrict__ W,
    const float* __restrict__ bias, const float* __restrict__ res,
    float* __restrict__ C, int N, int K) {
    constexpr int LDS20 = 20;                 // dword stride (40 bf16 = 80B, padded)
    __shared__ unsigned sA[128 * LDS20];
    __shared__ unsigned sW[128 * LDS20];

    const int tid  = threadIdx.x;
    const int lane = tid & 31;
    const int wid  = tid >> 5;
    const int wm   = wid >> 2;                // 0..1  (64-row slab)
    const int wn   = wid & 3;                 // 0..3  (32-col slab)
    const int lr   = lane & 15;
    const int lh   = lane >> 4;

    const int m0 = blockIdx.y * 128;
    const int n0 = blockIdx.x * 128;

    f32x8 acc[4][2];
#pragma unroll
    for (int mi = 0; mi < 4; ++mi)
#pragma unroll
        for (int ni = 0; ni < 2; ++ni)
#pragma unroll
            for (int r = 0; r < 8; ++r) acc[mi][ni][r] = 0.0f;

    for (int k0 = 0; k0 < K; k0 += 32) {
        __syncthreads();                      // previous tile fully consumed
#if HAVE_TDM
        if (tid < 32) {                       // wave 0 drives the Tensor Data Mover
            tdm_load_tile(A, K, (unsigned)m0, k0, sA);
            tdm_load_tile(W, K, (unsigned)n0, k0, sW);
            wait_tensorcnt0();
        }
#else
        if (k0 + 32 < K) {
            __builtin_prefetch(A + (size_t)m0 * K + k0 + 32, 0, 1);
            __builtin_prefetch(W + (size_t)n0 * K + k0 + 32, 0, 1);
        }
#pragma unroll
        for (int p = 0; p < 2; ++p) {
            int idx = p * 256 + tid;          // 0..511
            int r   = idx >> 2;               // 0..127
            int seg = idx & 3;                // 16B segment within 64B row
            uint4 va = *(const uint4*)(A + (size_t)(m0 + r) * K + k0 + seg * 8);
            *(uint4*)((char*)sA + r * 80 + seg * 16) = va;
            uint4 vw = *(const uint4*)(W + (size_t)(n0 + r) * K + k0 + seg * 8);
            *(uint4*)((char*)sW + r * 80 + seg * 16) = vw;
        }
#endif
        __syncthreads();

        bf16x16 af[4], wf[2];
#pragma unroll
        for (int mi = 0; mi < 4; ++mi)
            af[mi] = load_frag(sA, (wm * 64 + mi * 16 + lr) * LDS20, lh);
#pragma unroll
        for (int ni = 0; ni < 2; ++ni)
            wf[ni] = load_frag(sW, (wn * 32 + ni * 16 + lr) * LDS20, lh);

#pragma unroll
        for (int mi = 0; mi < 4; ++mi)
#pragma unroll
            for (int ni = 0; ni < 2; ++ni)
                acc[mi][ni] = __builtin_amdgcn_wmma_f32_16x16x32_bf16(
                    false, af[mi], false, wf[ni], (short)0, acc[mi][ni], false, false);
    }

    // C/D layout: lanes 0-15 -> N=lane, M=r ; lanes 16-31 -> N=lane-16, M=r+8
#pragma unroll
    for (int mi = 0; mi < 4; ++mi)
#pragma unroll
        for (int ni = 0; ni < 2; ++ni)
#pragma unroll
            for (int r = 0; r < 8; ++r) {
                int row = m0 + wm * 64 + mi * 16 + lh * 8 + r;
                int col = n0 + wn * 32 + ni * 16 + lr;
                float v = acc[mi][ni][r];
                if (HAS_BIAS) v += bias[col];
                if (ADD_RES) v += res[(size_t)row * N + col];
                C[(size_t)row * N + col] = v;
            }
}

// ---------------- fp32 -> packed bf16 conversion ----------------
__global__ __launch_bounds__(256) void f2bf_k(const float* __restrict__ in,
                                              unsigned* __restrict__ out, long long n2) {
    long long i = (long long)blockIdx.x * 256 + threadIdx.x;
    if (i < n2) {
        float2 v = ((const float2*)in)[i];
        out[i] = pack_bf2(v.x, v.y);
    }
}

// ---------------- LayerNorm (block per row) ----------------
template <bool BF16_OUT>
__global__ __launch_bounds__(256) void layernorm_k(
    const float* __restrict__ x, const float* __restrict__ g,
    const float* __restrict__ b, float* __restrict__ y,
    unsigned short* __restrict__ y16, int D) {
    const int row = blockIdx.x;
    const float* xr = x + (size_t)row * D;
    __shared__ float rs[256], rs2[256];
    float s = 0.f, s2 = 0.f;
    for (int i = threadIdx.x; i < D; i += 256) { float v = xr[i]; s += v; s2 += v * v; }
    rs[threadIdx.x] = s; rs2[threadIdx.x] = s2; __syncthreads();
    for (int off = 128; off > 0; off >>= 1) {
        if (threadIdx.x < off) { rs[threadIdx.x] += rs[threadIdx.x + off]; rs2[threadIdx.x] += rs2[threadIdx.x + off]; }
        __syncthreads();
    }
    float mu = rs[0] / D;
    float var = rs2[0] / D - mu * mu;
    float rinv = rsqrtf(var + 1e-5f);
    for (int i = threadIdx.x; i < D; i += 256) {
        float v = (xr[i] - mu) * rinv * g[i] + b[i];
        if (BF16_OUT) y16[(size_t)row * D + i] = f2bf1(v);
        else          y[(size_t)row * D + i] = v;
    }
}

// ---------------- depthwise causal conv (DC=4) + SiLU (fp32 + bf16 out) ----
__global__ __launch_bounds__(256) void dwconv_silu_k(
    const float* __restrict__ xz, const float* __restrict__ cw,
    const float* __restrict__ cb, float* __restrict__ xc,
    unsigned short* __restrict__ xc16) {
    size_t idx = (size_t)blockIdx.x * 256 + threadIdx.x;
    if (idx >= (size_t)kNT * kDI) return;
    int i = (int)(idx % kDI);
    size_t row = idx / kDI;
    int t = (int)(row % kT);
    float acc = cb[i];
#pragma unroll
    for (int j = 0; j < kDC; ++j) {
        int tj = t - (kDC - 1) + j;
        if (tj >= 0)
            acc += xz[(row - (kDC - 1) + j) * (size_t)(2 * kDI) + i] * cw[i * kDC + j];
    }
    float v = acc / (1.f + __expf(-acc));
    xc[idx] = v;
    xc16[idx] = f2bf1(v);
}

// ---------------- softplus + row mean -> dm ----------------
__global__ __launch_bounds__(256) void softplus_mean_k(
    const float* __restrict__ pre, float* __restrict__ dm, int D) {
    const int row = blockIdx.x;
    __shared__ float red[256];
    float s = 0.f;
    for (int i = threadIdx.x; i < D; i += 256) {
        float v = pre[(size_t)row * D + i];
        s += (v > 20.f) ? v : log1pf(__expf(v));
    }
    red[threadIdx.x] = s; __syncthreads();
    for (int off = 128; off > 0; off >>= 1) {
        if (threadIdx.x < off) red[threadIdx.x] += red[threadIdx.x + off];
        __syncthreads();
    }
    if (threadIdx.x == 0) dm[row] = red[0] / D;
}

// ---------------- SSD scan: 4 lanes per channel, 16 states each -------------
// grid (kDI/64, kB), 256 threads = 64 channels/block.
__global__ __launch_bounds__(256) void ssd_scan_k(
    const float* __restrict__ xc, const float* __restrict__ xz,
    const float* __restrict__ bc, const float* __restrict__ dmv,
    const float* __restrict__ Alog, const float* __restrict__ Dp,
    unsigned short* __restrict__ y16) {
    const int b  = blockIdx.y;
    const int ch = blockIdx.x * 64 + (threadIdx.x >> 2);
    const int q  = threadIdx.x & 3;
    const int s0 = q * 16;

    float Aneg[16], h[16];
#pragma unroll
    for (int s = 0; s < 16; ++s) {
        Aneg[s] = -__expf(Alog[(size_t)ch * kS + s0 + s]);
        h[s] = 0.f;
    }
    const float Dpi = Dp[ch];

    __shared__ float sB[kS], sC[kS], sdm[1];
    for (int t = 0; t < kT; ++t) {
        const int n = b * kT + t;
        __syncthreads();
        if (threadIdx.x < kS)           sB[threadIdx.x] = bc[(size_t)n * (2 * kS) + threadIdx.x];
        else if (threadIdx.x < 2 * kS)  sC[threadIdx.x - kS] = bc[(size_t)n * (2 * kS) + threadIdx.x];
        else if (threadIdx.x == 2 * kS) sdm[0] = dmv[n];
        __syncthreads();

        const float x  = xc[(size_t)n * kDI + ch];
        const float d  = sdm[0];
        const float xd = x * d;
        float acc = 0.f;
#pragma unroll
        for (int s = 0; s < 16; ++s) {
            float hs = __expf(d * Aneg[s]) * h[s] + xd * sB[s0 + s];
            h[s] = hs;
            acc += hs * sC[s0 + s];
        }
        acc += __shfl_xor(acc, 1, 32);
        acc += __shfl_xor(acc, 2, 32);
        if (q == 0) {
            const float zv = xz[(size_t)n * (2 * kDI) + kDI + ch];
            const float sz = zv / (1.f + __expf(-zv));
            y16[(size_t)n * kDI + ch] = f2bf1((acc + Dpi * x) * sz);
        }
    }
}

// ---------------- attention-pool head ----------------
__global__ __launch_bounds__(256) void rowdot_k(
    const float* __restrict__ x, const float* __restrict__ w,
    const float* __restrict__ b0, float* __restrict__ out, int D) {
    const int row = blockIdx.x;
    __shared__ float red[256];
    float s = 0.f;
    for (int i = threadIdx.x; i < D; i += 256) s += x[(size_t)row * D + i] * w[i];
    red[threadIdx.x] = s; __syncthreads();
    for (int off = 128; off > 0; off >>= 1) {
        if (threadIdx.x < off) red[threadIdx.x] += red[threadIdx.x + off];
        __syncthreads();
    }
    if (threadIdx.x == 0) out[row] = red[0] + b0[0];
}

__global__ __launch_bounds__(256) void softmax_stats_k(
    const float* __restrict__ logits, float* __restrict__ amax, float* __restrict__ asum) {
    const int b = blockIdx.x;
    __shared__ float red[256];
    float m = -1e30f;
    for (int t = threadIdx.x; t < kT; t += 256) m = fmaxf(m, logits[b * kT + t]);
    red[threadIdx.x] = m; __syncthreads();
    for (int off = 128; off > 0; off >>= 1) {
        if (threadIdx.x < off) red[threadIdx.x] = fmaxf(red[threadIdx.x], red[threadIdx.x + off]);
        __syncthreads();
    }
    const float mx = red[0]; __syncthreads();
    float s = 0.f;
    for (int t = threadIdx.x; t < kT; t += 256) s += __expf(logits[b * kT + t] - mx);
    red[threadIdx.x] = s; __syncthreads();
    for (int off = 128; off > 0; off >>= 1) {
        if (threadIdx.x < off) red[threadIdx.x] += red[threadIdx.x + off];
        __syncthreads();
    }
    if (threadIdx.x == 0) { amax[b] = mx; asum[b] = red[0]; }
}

__global__ __launch_bounds__(256) void pool_k(
    const float* __restrict__ hf, const float* __restrict__ logits,
    const float* __restrict__ amax, const float* __restrict__ asum,
    float* __restrict__ pooled) {
    const int b = blockIdx.x;
    __shared__ float w[kT];
    const float mx = amax[b], inv = 1.f / asum[b];
    for (int t = threadIdx.x; t < kT; t += 256) w[t] = __expf(logits[b * kT + t] - mx) * inv;
    __syncthreads();
    for (int d = threadIdx.x; d < kDM; d += 256) {
        float acc = 0.f;
        for (int t = 0; t < kT; ++t) acc += hf[((size_t)(b * kT + t)) * kDM + d] * w[t];
        pooled[(size_t)b * kDM + d] = acc;
    }
}

__global__ __launch_bounds__(64) void ctx_k(
    const float* __restrict__ pooled, const float* __restrict__ Wc,
    const float* __restrict__ bct, const float* __restrict__ g,
    const float* __restrict__ bb, float* __restrict__ out) {
    const int b = blockIdx.x;
    const int j = threadIdx.x;  // 64
    const float* p = pooled + (size_t)b * kDM;
    float acc = bct[j];
    for (int k = 0; k < kDM; ++k) acc += p[k] * Wc[(size_t)j * kDM + k];
    __shared__ float sv[64];
    sv[j] = acc; __syncthreads();
    float s = 0.f, s2 = 0.f;
    for (int k = 0; k < 64; ++k) { s += sv[k]; s2 += sv[k] * sv[k]; }
    float mu = s / 64.f, var = s2 / 64.f - mu * mu;
    float v = (acc - mu) * rsqrtf(var + 1e-5f) * g[j] + bb[j];
    out[(size_t)b * 64 + j] = 0.5f * v * (1.f + erff(v * 0.70710678f));
}

// ---------------- host-side orchestration ----------------
extern "C" void kernel_launch(void* const* d_in, const int* in_sizes, int n_in,
                              void* d_out, int out_size, void* d_ws, size_t ws_size,
                              hipStream_t stream) {
    (void)in_sizes; (void)n_in; (void)out_size; (void)ws_size;
    const float* x        = (const float*)d_in[0];
    const float* W_inp    = (const float*)d_in[1];
    const float* b_inp    = (const float*)d_in[2];
    const float* in_ln_g  = (const float*)d_in[3];
    const float* in_ln_b  = (const float*)d_in[4];
    const float* ln_g     = (const float*)d_in[5];
    const float* ln_b     = (const float*)d_in[6];
    const float* W_in     = (const float*)d_in[7];
    const float* conv_w   = (const float*)d_in[8];
    const float* conv_b   = (const float*)d_in[9];
    const float* W_xbc    = (const float*)d_in[10];
    const float* W_dt     = (const float*)d_in[11];
    const float* b_dt     = (const float*)d_in[12];
    const float* A_log    = (const float*)d_in[13];
    const float* Dp       = (const float*)d_in[14];
    const float* W_out    = (const float*)d_in[15];
    const float* out_ln_g = (const float*)d_in[16];
    const float* out_ln_b = (const float*)d_in[17];
    const float* W_pool   = (const float*)d_in[18];
    const float* b_pool   = (const float*)d_in[19];
    const float* W_ctx    = (const float*)d_in[20];
    const float* b_ctx    = (const float*)d_in[21];
    const float* ctx_ln_g = (const float*)d_in[22];
    const float* ctx_ln_b = (const float*)d_in[23];
    float* out = (float*)d_out;

    // ---- workspace carve-up (256B aligned blocks) ----
    char* wp = (char*)d_ws;
    auto alloc = [&](size_t bytes) { void* r = (void*)wp; wp += (bytes + 255) & ~(size_t)255; return r; };
    float* hA     = (float*)alloc((size_t)kNT * kDM * 4);
    float* hB     = (float*)alloc((size_t)kNT * kDM * 4);
    float* xzb    = (float*)alloc((size_t)kNT * 2 * kDI * 4);
    float* xcb    = (float*)alloc((size_t)kNT * kDI * 4);
    float* dtpre  = (float*)alloc((size_t)kNT * kDI * 4);
    float* bcb    = (float*)alloc((size_t)kNT * 2 * kS * 4);
    float* dmv    = (float*)alloc((size_t)kNT * 4);
    float* logits = (float*)alloc((size_t)kNT * 4);
    float* amax   = (float*)alloc((size_t)kB * 4);
    float* asum   = (float*)alloc((size_t)kB * 4);
    float* pooled = (float*)alloc((size_t)kB * kDM * 4);
    unsigned short* x16    = (unsigned short*)alloc((size_t)kNT * kDIN * 2);
    unsigned short* xn16   = (unsigned short*)alloc((size_t)kNT * kDM * 2);
    unsigned short* xc16   = (unsigned short*)alloc((size_t)kNT * kDI * 2);
    unsigned short* y16    = (unsigned short*)alloc((size_t)kNT * kDI * 2);
    unsigned short* wInp16 = (unsigned short*)alloc((size_t)kDM * kDIN * 2);
    unsigned short* wIn16  = (unsigned short*)alloc((size_t)kL * 2 * kDI * kDM * 2);
    unsigned short* wXbc16 = (unsigned short*)alloc((size_t)kL * 2 * kS * kDI * 2);
    unsigned short* wDt16  = (unsigned short*)alloc((size_t)kL * kDI * kDI * 2);
    unsigned short* wOut16 = (unsigned short*)alloc((size_t)kL * kDM * kDI * 2);

    const dim3 blk(256);
    auto cvt = [&](const float* src, unsigned short* dst, size_t n) {
        long long n2 = (long long)(n / 2);
        f2bf_k<<<dim3((unsigned)((n2 + 255) / 256)), blk, 0, stream>>>(src, (unsigned*)dst, n2);
    };

    // ---- one-time bf16 packing of GEMM operands ----
    cvt(x, x16, (size_t)kNT * kDIN);
    cvt(W_inp, wInp16, (size_t)kDM * kDIN);
    cvt(W_in, wIn16, (size_t)kL * 2 * kDI * kDM);
    cvt(W_xbc, wXbc16, (size_t)kL * 2 * kS * kDI);
    cvt(W_dt, wDt16, (size_t)kL * kDI * kDI);
    cvt(W_out, wOut16, (size_t)kL * kDM * kDI);

    const dim3 gIn(kDM / 128, kNT / 128);        // (6, 64)
    const dim3 gXZ(2 * kDI / 128, kNT / 128);    // (24, 64)
    const dim3 gBC(2 * kS / 128, kNT / 128);     // (1, 64)
    const dim3 gDT(kDI / 128, kNT / 128);        // (12, 64)
    const dim3 gOut(kDM / 128, kNT / 128);       // (6, 64)
    const int nConvBlk = (int)(((size_t)kNT * kDI + 255) / 256);

    // input projection + input LN
    gemm_bf16<true, false><<<gIn, blk, 0, stream>>>(x16, wInp16, b_inp, nullptr, hB, kDM, kDIN);
    layernorm_k<false><<<kNT, blk, 0, stream>>>(hB, in_ln_g, in_ln_b, hA, nullptr, kDM);

    float* cur = hA;
    float* tmp = hB;
    for (int l = 0; l < kL; ++l) {
        layernorm_k<true><<<kNT, blk, 0, stream>>>(cur, ln_g + (size_t)l * kDM, ln_b + (size_t)l * kDM,
                                                   nullptr, xn16, kDM);
        gemm_bf16<false, false><<<gXZ, blk, 0, stream>>>(
            xn16, wIn16 + (size_t)l * 2 * kDI * kDM, nullptr, nullptr, xzb, 2 * kDI, kDM);
        dwconv_silu_k<<<nConvBlk, blk, 0, stream>>>(
            xzb, conv_w + (size_t)l * kDI * kDC, conv_b + (size_t)l * kDI, xcb, xc16);
        gemm_bf16<false, false><<<gBC, blk, 0, stream>>>(
            xc16, wXbc16 + (size_t)l * 2 * kS * kDI, nullptr, nullptr, bcb, 2 * kS, kDI);
        gemm_bf16<true, false><<<gDT, blk, 0, stream>>>(
            xc16, wDt16 + (size_t)l * kDI * kDI, b_dt + (size_t)l * kDI, nullptr, dtpre, kDI, kDI);
        softplus_mean_k<<<kNT, blk, 0, stream>>>(dtpre, dmv, kDI);
        ssd_scan_k<<<dim3(kDI / 64, kB), blk, 0, stream>>>(
            xcb, xzb, bcb, dmv, A_log + (size_t)l * kDI * kS, Dp + (size_t)l * kDI, y16);
        gemm_bf16<false, true><<<gOut, blk, 0, stream>>>(
            y16, wOut16 + (size_t)l * kDM * kDI, nullptr, cur, tmp, kDM, kDI);
        std::swap(cur, tmp);
    }

    // final LN + attention pooling + context head
    layernorm_k<false><<<kNT, blk, 0, stream>>>(cur, out_ln_g, out_ln_b, tmp, nullptr, kDM);
    rowdot_k<<<kNT, blk, 0, stream>>>(tmp, W_pool, b_pool, logits, kDM);
    softmax_stats_k<<<kB, blk, 0, stream>>>(logits, amax, asum);
    pool_k<<<kB, blk, 0, stream>>>(tmp, logits, amax, asum, pooled);
    ctx_k<<<kB, dim3(64), 0, stream>>>(pooled, W_ctx, b_ctx, ctx_ln_g, ctx_ln_b, out);
}
